// VSSBlock_48782238548074
// MI455X (gfx1250) — matrix operand order repaired and checked by reading, not compiled, not run on hardware
//
#include <hip/hip_runtime.h>
#include <hip/hip_bf16.h>
#include <math.h>

// ---------------- problem constants (from reference) ----------------
#define Bv   4
#define Hh   64
#define Ww   64
#define Cc   96
#define Dm   192          // D_INNER
#define Kd   4            // directions
#define Nn   16           // state size
#define Rr   6            // DT_RANK
#define Ll   4096         // H*W
#define Cdbl 38           // DT_RANK + 2N
#define XPR  48           // x_proj weight rows padded 38 -> 48

typedef __attribute__((ext_vector_type(16))) _Float16 v16h;
typedef __attribute__((ext_vector_type(8)))  _Float16 v8h;
typedef __attribute__((ext_vector_type(8)))  float    v8f;

// ---------------- WMMA fragment helpers (ISA 7.12.2 lane layouts) -------
__device__ __forceinline__ v16h cat8(v8h lo, v8h hi) {
    return __builtin_shufflevector(lo, hi, 0,1,2,3,4,5,6,7,8,9,10,11,12,13,14,15);
}
// 16-bit A 16x32 (MxK): per lane, K runs [half*8,+7] and [16+half*8,+7] contiguous.
__device__ __forceinline__ v16h load_a16(const _Float16* rowp, int half) {
    v8h lo = *(const v8h*)(rowp + half * 8);
    v8h hi = *(const v8h*)(rowp + 16 + half * 8);
    return cat8(lo, hi);
}
// 16-bit B 32x16 (KxN): per lane (column n), K runs [half*8,+7] and [16+half*8,+7]
// contiguous in an N-major (transposed) buffer.
__device__ __forceinline__ v16h load_b16(const _Float16* colp, int half) {
    v8h lo = *(const v8h*)(colp + half * 8);
    v8h hi = *(const v8h*)(colp + 16 + half * 8);
    return cat8(lo, hi);
}
// 32-bit C/D 16x16: VGPR r -> row r (lanes 0-15) / row 8+r (lanes 16-31)
__device__ __forceinline__ void cd_coords(int lane, int r, int& m, int& n) {
    m = r + ((lane >> 4) << 3);
    n = lane & 15;
}

// spatial index into row-major conv output for scan direction k, position l
__device__ __forceinline__ int xs_index(int k, int l) {
    int lp = (k >= 2) ? (Ll - 1 - l) : l;       // flip for k=2,3
    if (k & 1) {                                // transposed scan (w-major)
        int wq = lp / Hh, hq = lp % Hh;
        return hq * Ww + wq;
    }
    return lp;
}

// ===== kernel W: one-shot weight conversion to f16 (B-side transposed) ===
__global__ void k_cvt_w(const float* __restrict__ ipw, const float* __restrict__ opw,
                        const float* __restrict__ xpw, const float* __restrict__ dtw,
                        _Float16* __restrict__ wT_ip,   // (2Dm, Cc)   in_proj^T
                        _Float16* __restrict__ wT_op,   // (Cc, Dm)    out_proj^T
                        _Float16* __restrict__ xpw_h,   // (Kd,48,Dm)  zero-padded rows
                        _Float16* __restrict__ dtw_h) { // (Kd,Dm,32)  zero-padded K
    int i = blockIdx.x * blockDim.x + threadIdx.x;
    if (i < 2 * Dm * Cc) {                      // transpose (Cc,2Dm)->(2Dm,Cc)
        int n = i / Cc, k = i % Cc;
        wT_ip[i] = (_Float16)ipw[(size_t)k * (2 * Dm) + n];
    }
    if (i < Cc * Dm) {                          // transpose (Dm,Cc)->(Cc,Dm)
        int n = i / Dm, k = i % Dm;
        wT_op[i] = (_Float16)opw[(size_t)k * Cc + n];
    }
    if (i < Kd * XPR * Dm) {                    // pad 38 rows -> 48 with zeros
        int dcol = i % Dm; int t = i / Dm;
        int row = t % XPR, kq = t / XPR;
        xpw_h[i] = (row < Cdbl) ? (_Float16)xpw[((size_t)kq * Cdbl + row) * Dm + dcol]
                                : (_Float16)0.f;
    }
    if (i < Kd * Dm * 32) {                     // pad DT_RANK=6 -> 32 with zeros
        int r = i & 31, t = i >> 5;
        int d = t % Dm, kq = t / Dm;
        dtw_h[i] = (r < Rr) ? (_Float16)dtw[((size_t)kq * Dm + d) * Rr + r] : (_Float16)0.f;
    }
}

// ============ kernel 0: LayerNorm over C=96, emit f16 activations ========
__global__ void k_ln(const float* __restrict__ x, const float* __restrict__ g,
                     const float* __restrict__ b, _Float16* __restrict__ xn) {
    int wave = (blockIdx.x * blockDim.x + threadIdx.x) >> 5;   // one wave / pixel
    int lane = threadIdx.x & 31;
    if (wave >= Bv * Ll) return;
    const float* row = x + (size_t)wave * Cc;
    float v[3], s = 0.f, ss = 0.f;
    #pragma unroll
    for (int i = 0; i < 3; ++i) {                              // 96 = 32 lanes * 3
        v[i] = row[lane * 3 + i];
        s += v[i]; ss += v[i] * v[i];
    }
    #pragma unroll
    for (int off = 16; off; off >>= 1) {
        s  += __shfl_xor(s,  off, 32);
        ss += __shfl_xor(ss, off, 32);
    }
    float mu = s * (1.f / Cc);
    float rs = rsqrtf(ss * (1.f / Cc) - mu * mu + 1e-6f);
    #pragma unroll
    for (int i = 0; i < 3; ++i) {
        int c = lane * 3 + i;
        xn[(size_t)wave * Cc + c] = (_Float16)(((v[i] - mu) * rs) * g[c] + b[c]);
    }
}

// ====== kernel 1: in_proj WMMA GEMM, 16x64 per wave (A reused 4x) ========
__global__ void k_inproj(const _Float16* __restrict__ xn, const _Float16* __restrict__ wT,
                         const float* __restrict__ bias,
                         float* __restrict__ xi_pre,   // (B, Dm, L)  conv input
                         float* __restrict__ z) {      // (B, L, Dm)  gate branch
    int wid  = (blockIdx.x * blockDim.x + threadIdx.x) >> 5;
    int lane = threadIdx.x & 31;
    const int MT = (Bv * Ll) / 16, NG = (2 * Dm) / 64;
    if (wid >= MT * NG) return;
    int mt = wid % MT, ng = wid / MT;
    int m0 = mt * 16, n0 = ng * 64;
    int half = lane >> 4, mn = lane & 15;
    const _Float16* arow = xn + (size_t)(m0 + mn) * Cc;
    const _Float16* brow = wT + (size_t)(n0 + mn) * Cc;
    v8f acc[4] = {};
    #pragma unroll
    for (int k0 = 0; k0 < Cc; k0 += 32) {
        v16h a = load_a16(arow + k0, half);
        #pragma unroll
        for (int t = 0; t < 4; ++t) {
            v16h bm = load_b16(brow + (size_t)t * 16 * Cc + k0, half);
            acc[t] = __builtin_amdgcn_wmma_f32_16x16x32_f16(false, a, false, bm,
                                                            (short)0, acc[t], false, false);
        }
    }
    int bb = m0 / Ll, l = m0 % Ll;               // tile never crosses batch rows
    if (n0 < Dm) {                               // uniform: whole group is xi half
        #pragma unroll
        for (int t = 0; t < 4; ++t)
            #pragma unroll
            for (int r = 0; r < 8; ++r) {
                int mm, nn; cd_coords(lane, r, mm, nn);
                int gn = n0 + t * 16 + nn;
                xi_pre[((size_t)bb * Dm + gn) * Ll + l + mm] = acc[t][r] + bias[gn];
            }
    } else {                                     // uniform: whole group is z half
        #pragma unroll
        for (int t = 0; t < 4; ++t)
            #pragma unroll
            for (int r = 0; r < 8; ++r) {
                int mm, nn; cd_coords(lane, r, mm, nn);
                int gn = n0 + t * 16 + nn;
                z[(size_t)(m0 + mm) * Dm + (gn - Dm)] = acc[t][r] + bias[gn];
            }
    }
}

// ============ kernel 2: depthwise 3x3 conv + bias + SiLU =================
// writes f32 (B,Dm,L) for the scan and f16 transposed (B,L,Dm) for x_proj
__global__ void k_conv(const float* __restrict__ xi, const float* __restrict__ cw,
                       const float* __restrict__ cb, float* __restrict__ xc,
                       _Float16* __restrict__ xcT) {
    size_t idx = (size_t)blockIdx.x * blockDim.x + threadIdx.x;
    const size_t total = (size_t)Bv * Dm * Ll;
    if (idx >= total) return;
    int l = (int)(idx % Ll);
    size_t t = idx / Ll;
    int dch = (int)(t % Dm);
    int bb  = (int)(t / Dm);
    int h = l / Ww, wcol = l % Ww;
    const float* wk  = cw + (size_t)dch * 9;
    const float* src = xi + ((size_t)bb * Dm + dch) * Ll;
    float s = 0.f;
    #pragma unroll
    for (int dy = -1; dy <= 1; ++dy)
        #pragma unroll
        for (int dx = -1; dx <= 1; ++dx) {
            int hy = h + dy, wx = wcol + dx;
            if (hy >= 0 && hy < Hh && wx >= 0 && wx < Ww)
                s += wk[(dy + 1) * 3 + (dx + 1)] * src[hy * Ww + wx];
        }
    s += cb[dch];
    float act = s / (1.f + __expf(-s));     // SiLU
    xc[idx] = act;
    xcT[((size_t)bb * Ll + l) * Dm + dch] = (_Float16)act;
}

// ====== kernel 3: x_proj WMMA GEMM, 48x16 per wave (B reused 3x) =========
__global__ void k_xproj(const _Float16* __restrict__ xpw_h, const _Float16* __restrict__ xcT,
                        float* __restrict__ xdbl,        // (B*K, 38, L)
                        _Float16* __restrict__ dtr_h) {  // (B*K, L, 8) dt_r, K-padded
    int wid  = (blockIdx.x * blockDim.x + threadIdx.x) >> 5;
    int lane = threadIdx.x & 31;
    const int NT = Ll / 16;
    if (wid >= Bv * Kd * NT) return;
    int nt = wid % NT, bk = wid / NT;
    int k = bk % Kd, bb = bk / Kd;
    int l0 = nt * 16;
    int half = lane >> 4, mn = lane & 15;
    int sp = xs_index(k, l0 + mn);                     // direction remap per column
    const _Float16* brow = xcT + ((size_t)bb * Ll + sp) * Dm;
    const _Float16* abase = xpw_h + (size_t)k * XPR * Dm + (size_t)mn * Dm;
    v8f acc[3] = {};
    #pragma unroll
    for (int k0 = 0; k0 < Dm; k0 += 32) {
        v16h bm = load_b16(brow + k0, half);
        #pragma unroll
        for (int t = 0; t < 3; ++t) {
            v16h a = load_a16(abase + (size_t)t * 16 * Dm + k0, half);  // padded rows
            acc[t] = __builtin_amdgcn_wmma_f32_16x16x32_f16(false, a, false, bm,
                                                            (short)0, acc[t], false, false);
        }
    }
    #pragma unroll
    for (int t = 0; t < 3; ++t)
        #pragma unroll
        for (int r = 0; r < 8; ++r) {
            int mm, nn; cd_coords(lane, r, mm, nn);
            int c = t * 16 + mm;
            if (t == 2 && c >= Cdbl) continue;          // only last tile masked
            xdbl[((size_t)bk * Cdbl + c) * Ll + l0 + nn] = acc[t][r];
        }
    // dt_r rows (0..5) as zero-padded f16 octets for the dt_proj B operand
    #pragma unroll
    for (int r = 0; r < 8; ++r) {
        int mm, nn; cd_coords(lane, r, mm, nn);
        size_t o = ((size_t)bk * Ll + l0 + nn) * 8;
        if (mm < Rr)              dtr_h[o + mm] = (_Float16)acc[0][r];
        else if (mm < 8)          dtr_h[o + mm] = (_Float16)0.f;   // K pad
    }
}

// ====== kernel 4: dt_proj WMMA GEMM, 192x16 per wave (B reused 12x) ======
__global__ void k_dtproj(const _Float16* __restrict__ dtw_h, const float* __restrict__ dtb,
                         const _Float16* __restrict__ dtr_h, float* __restrict__ dts) {
    int wid  = (blockIdx.x * blockDim.x + threadIdx.x) >> 5;
    int lane = threadIdx.x & 31;
    const int NT = Ll / 16;
    if (wid >= Bv * Kd * NT) return;
    int nt = wid % NT, bk = wid / NT;
    int k = bk % Kd;
    int l0 = nt * 16;
    int half = lane >> 4, mn = lane & 15;
    // B: column l0+mn of (6pad8, L) dt_r -> one 16B load, upper K all zero
    v16h bm;
    {
        v8h zz = {};
        if (half == 0) {
            v8h lo = *(const v8h*)(dtr_h + ((size_t)bk * Ll + l0 + mn) * 8);
            bm = cat8(lo, zz);                 // K=0..7 (6 real + 2 pad), K=16..23 zero
        } else {
            bm = cat8(zz, zz);                 // K=8..15 and 24..31 zero
        }
    }
    const _Float16* abase = dtw_h + ((size_t)k * Dm + mn) * 32;
    v8f acc[12] = {};
    #pragma unroll
    for (int t = 0; t < 12; ++t) {
        v16h a = load_a16(abase + (size_t)t * 16 * 32, half);
        acc[t] = __builtin_amdgcn_wmma_f32_16x16x32_f16(false, a, false, bm,
                                                        (short)0, acc[t], false, false);
    }
    #pragma unroll
    for (int t = 0; t < 12; ++t)
        #pragma unroll
        for (int r = 0; r < 8; ++r) {
            int mm, nn; cd_coords(lane, r, mm, nn);
            int d = t * 16 + mm;
            float val = acc[t][r] + dtb[k * Dm + d];
            float sp = (val > 20.f) ? val : log1pf(__expf(val));   // softplus
            dts[((size_t)bk * Dm + d) * Ll + l0 + nn] = sp;
        }
}

// ============ kernel 5: selective scan, 16 lanes (one per n) per (b,k,d) =
__global__ void k_scan(const float* __restrict__ dts, const float* __restrict__ xdbl,
                       const float* __restrict__ xc, const float* __restrict__ A_logs,
                       const float* __restrict__ Ds, float* __restrict__ ys) {
    int t = blockIdx.x * blockDim.x + threadIdx.x;
    int pair = t >> 4;                         // (b*K + k)*Dm + d
    int n = t & 15;
    if (pair >= Bv * Kd * Dm) return;
    int d  = pair % Dm;
    int bk = pair / Dm;
    int k  = bk % Kd, bb = bk / Kd;
    float An = -__expf(A_logs[(size_t)(k * Dm + d) * Nn + n]);
    float Dv = Ds[k * Dm + d];
    const float* dtp = dts  + ((size_t)bk * Dm + d) * Ll;
    const float* Bp  = xdbl + ((size_t)bk * Cdbl + Rr + n) * Ll;
    const float* Cp  = xdbl + ((size_t)bk * Cdbl + Rr + Nn + n) * Ll;
    const float* up  = xc   + ((size_t)bb * Dm + d) * Ll;
    float* yp        = ys   + ((size_t)bk * Dm + d) * Ll;
    float h = 0.f;
    for (int l = 0; l < Ll; ++l) {
        float dt = dtp[l];
        float u  = up[xs_index(k, l)];
        float dA = __expf(dt * An);
        h = dA * h + (dt * u) * Bp[l];
        float p = h * Cp[l];
        p += __shfl_xor(p, 1, 32);             // reduce over the 16-lane n-group
        p += __shfl_xor(p, 2, 32);
        p += __shfl_xor(p, 4, 32);
        p += __shfl_xor(p, 8, 32);
        if (n == 0) yp[l] = p + Dv * u;        // + D * u skip term
    }
}

// ============ kernel 6: gather 4 dirs + out LN + SiLU gate -> f16 ========
__global__ void k_gather(const float* __restrict__ ys, const float* __restrict__ z,
                         const float* __restrict__ g, const float* __restrict__ b,
                         _Float16* __restrict__ gate) {
    int wave = (blockIdx.x * blockDim.x + threadIdx.x) >> 5;
    int lane = threadIdx.x & 31;
    if (wave >= Bv * Ll) return;
    int bb = wave / Ll, l = wave % Ll;
    int h = l / Ww, wcol = l % Ww;
    int lT = wcol * Hh + h;                    // transposed position
    float v[6], s = 0.f, ss = 0.f;
    #pragma unroll
    for (int i = 0; i < 6; ++i) {              // 192 = 32 lanes * 6
        int d = lane * 6 + i;
        float y0 = ys[(((size_t)bb * Kd + 0) * Dm + d) * Ll + l];
        float y1 = ys[(((size_t)bb * Kd + 1) * Dm + d) * Ll + lT];
        float y2 = ys[(((size_t)bb * Kd + 2) * Dm + d) * Ll + (Ll - 1 - l)];
        float y3 = ys[(((size_t)bb * Kd + 3) * Dm + d) * Ll + (Ll - 1 - lT)];
        v[i] = y0 + y1 + y2 + y3;
        s += v[i]; ss += v[i] * v[i];
    }
    #pragma unroll
    for (int off = 16; off; off >>= 1) {
        s  += __shfl_xor(s,  off, 32);
        ss += __shfl_xor(ss, off, 32);
    }
    float mu = s * (1.f / Dm);
    float rs = rsqrtf(ss * (1.f / Dm) - mu * mu + 1e-5f);
    #pragma unroll
    for (int i = 0; i < 6; ++i) {
        int d = lane * 6 + i;
        float yn = (v[i] - mu) * rs * g[d] + b[d];
        float zz = z[(size_t)wave * Dm + d];
        float gz = zz / (1.f + __expf(-zz));   // silu(z)
        gate[(size_t)wave * Dm + d] = (_Float16)(yn * gz);
    }
}

// ====== kernel 7: out_proj WMMA GEMM, 16x96 per wave (A reused 6x) =======
__global__ void k_outproj(const _Float16* __restrict__ gate, const _Float16* __restrict__ wT,
                          const float* __restrict__ bias, const float* __restrict__ x,
                          float* __restrict__ out) {
    int wid  = (blockIdx.x * blockDim.x + threadIdx.x) >> 5;
    int lane = threadIdx.x & 31;
    const int MT = (Bv * Ll) / 16;
    if (wid >= MT) return;
    int m0 = wid * 16;
    int half = lane >> 4, mn = lane & 15;
    const _Float16* arow = gate + (size_t)(m0 + mn) * Dm;
    const _Float16* brow = wT   + (size_t)mn * Dm;
    v8f acc[6] = {};
    #pragma unroll
    for (int k0 = 0; k0 < Dm; k0 += 32) {
        v16h a = load_a16(arow + k0, half);
        #pragma unroll
        for (int t = 0; t < 6; ++t) {
            v16h bm = load_b16(brow + (size_t)t * 16 * Dm + k0, half);
            acc[t] = __builtin_amdgcn_wmma_f32_16x16x32_f16(false, a, false, bm,
                                                            (short)0, acc[t], false, false);
        }
    }
    #pragma unroll
    for (int t = 0; t < 6; ++t)
        #pragma unroll
        for (int r = 0; r < 8; ++r) {
            int mm, nn; cd_coords(lane, r, mm, nn);
            int gn = t * 16 + nn;
            size_t o = (size_t)(m0 + mm) * Cc + gn;
            out[o] = acc[t][r] + bias[gn] + x[o];    // residual add
        }
}

// ---------------- host orchestration ------------------------------------
static inline size_t align256(size_t v) { return (v + 255) & ~(size_t)255; }
static inline int cdiv(long a, long b) { return (int)((a + b - 1) / b); }

extern "C" void kernel_launch(void* const* d_in, const int* in_sizes, int n_in,
                              void* d_out, int out_size, void* d_ws, size_t ws_size,
                              hipStream_t stream) {
    (void)in_sizes; (void)n_in; (void)out_size; (void)ws_size;
    const float* x      = (const float*)d_in[0];
    const float* ln_g   = (const float*)d_in[1];
    const float* ln_b   = (const float*)d_in[2];
    const float* ipw    = (const float*)d_in[3];
    const float* ipb    = (const float*)d_in[4];
    const float* cw     = (const float*)d_in[5];
    const float* cb     = (const float*)d_in[6];
    const float* xpw    = (const float*)d_in[7];
    const float* dtw    = (const float*)d_in[8];
    const float* dtb    = (const float*)d_in[9];
    const float* A_logs = (const float*)d_in[10];
    const float* Ds     = (const float*)d_in[11];
    const float* ong    = (const float*)d_in[12];
    const float* onb    = (const float*)d_in[13];
    const float* opw    = (const float*)d_in[14];
    const float* opb    = (const float*)d_in[15];
    float* out = (float*)d_out;

    // workspace carve-up
    char* ws = (char*)d_ws;
    size_t off = 0;
    _Float16* xn_h  = (_Float16*)(ws + off); off += align256((size_t)Bv * Ll * Cc * sizeof(_Float16));
    float* xi_pre   = (float*)(ws + off);    off += align256((size_t)Bv * Dm * Ll * sizeof(float));
    float* zbuf     = (float*)(ws + off);    off += align256((size_t)Bv * Ll * Dm * sizeof(float));
    float* xc       = (float*)(ws + off);    off += align256((size_t)Bv * Dm * Ll * sizeof(float));
    _Float16* xcT   = (_Float16*)(ws + off); off += align256((size_t)Bv * Ll * Dm * sizeof(_Float16));
    float* xdbl     = (float*)(ws + off);    off += align256((size_t)Bv * Kd * Cdbl * Ll * sizeof(float));
    _Float16* dtr_h = (_Float16*)(ws + off); off += align256((size_t)Bv * Kd * Ll * 8 * sizeof(_Float16));
    float* dts      = (float*)(ws + off);    off += align256((size_t)Bv * Kd * Dm * Ll * sizeof(float));
    float* ys       = (float*)(ws + off);    off += align256((size_t)Bv * Kd * Dm * Ll * sizeof(float));
    _Float16* gate  = (_Float16*)(ws + off); off += align256((size_t)Bv * Ll * Dm * sizeof(_Float16));
    _Float16* wT_ip = (_Float16*)(ws + off); off += align256((size_t)2 * Dm * Cc * sizeof(_Float16));
    _Float16* wT_op = (_Float16*)(ws + off); off += align256((size_t)Cc * Dm * sizeof(_Float16));
    _Float16* xpw_h = (_Float16*)(ws + off); off += align256((size_t)Kd * XPR * Dm * sizeof(_Float16));
    _Float16* dtw_h = (_Float16*)(ws + off); off += align256((size_t)Kd * Dm * 32 * sizeof(_Float16));

    const int BLK = 256;   // 8 wave32s

    // W: weight conversion (one shot; grid covers the largest matrix: 36864)
    k_cvt_w<<<cdiv(2L * Dm * Cc, BLK), BLK, 0, stream>>>(ipw, opw, xpw, dtw,
                                                         wT_ip, wT_op, xpw_h, dtw_h);

    // 0: input LayerNorm (wave per pixel)
    k_ln<<<cdiv((long)Bv * Ll * 32, BLK), BLK, 0, stream>>>(x, ln_g, ln_b, xn_h);

    // 1: in_proj WMMA GEMM (1024 M-tiles x 6 N-groups of 64)
    {
        long waves = ((long)(Bv * Ll) / 16) * ((2 * Dm) / 64);
        k_inproj<<<cdiv(waves * 32, BLK), BLK, 0, stream>>>(xn_h, wT_ip, ipb, xi_pre, zbuf);
    }

    // 2: depthwise conv + SiLU (f32 for scan, f16^T for x_proj)
    k_conv<<<cdiv((long)Bv * Dm * Ll, BLK), BLK, 0, stream>>>(xi_pre, cw, cb, xc, xcT);

    // 3: x_proj WMMA GEMM per (b,k): 48 rows x 16 cols per wave
    {
        long waves = (long)Bv * Kd * (Ll / 16);
        k_xproj<<<cdiv(waves * 32, BLK), BLK, 0, stream>>>(xpw_h, xcT, xdbl, dtr_h);
    }

    // 4: dt_proj WMMA GEMM + softplus: 192 rows x 16 cols per wave
    {
        long waves = (long)Bv * Kd * (Ll / 16);
        k_dtproj<<<cdiv(waves * 32, BLK), BLK, 0, stream>>>(dtw_h, dtb, dtr_h, dts);
    }

    // 5: selective scan (16 lanes per (b,k,d) channel)
    k_scan<<<cdiv((long)Bv * Kd * Dm * 16, BLK), BLK, 0, stream>>>(dts, xdbl, xc, A_logs, Ds, ys);

    // 6: direction gather + out LayerNorm + SiLU gate
    k_gather<<<cdiv((long)Bv * Ll * 32, BLK), BLK, 0, stream>>>(ys, zbuf, ong, onb, gate);

    // 7: out_proj WMMA GEMM + bias + residual: 16x96 per wave
    {
        long waves = (long)(Bv * Ll) / 16;
        k_outproj<<<cdiv(waves * 32, BLK), BLK, 0, stream>>>(gate, wT_op, opb, x, out);
    }
}
